// Net_3487513444357
// MI455X (gfx1250) — matrix operand order repaired and checked by reading, not compiled, not run on hardware
//
#include <hip/hip_runtime.h>

#define NN 16384
#define EE 65536
#define GG 128
#define DD 64
#define LL 3
#define EPSV 1e-5f

typedef __attribute__((ext_vector_type(16))) __bf16 v16bf;
typedef __attribute__((ext_vector_type(8)))  float  v8f;

#if defined(__gfx1250__) && __has_builtin(__builtin_amdgcn_global_load_async_to_lds_b128) && __has_builtin(__builtin_amdgcn_s_wait_asynccnt)
#define USE_ASYNC 1
#else
#define USE_ASYNC 0
#endif

// ---------------------------------------------------------------- helpers
__device__ __forceinline__ unsigned f32_ord(float f) {
  unsigned i = __float_as_uint(f);
  return (i & 0x80000000u) ? ~i : (i | 0x80000000u);
}
__device__ __forceinline__ float ord_f32(unsigned k) {
  unsigned i = (k & 0x80000000u) ? (k & 0x7fffffffu) : ~k;
  return __uint_as_float(i);
}

#if USE_ASYNC
// 32-byte async global->LDS copy (2x b128) on the gfx1250 async engine (ASYNCcnt).
// Builtin signature (from compiler diagnostic): arg0 = generic `int4*` (vector_size(16))
// source address, arg1 = LDS destination, then imm offset + imm cpol.
typedef int v4i_vs __attribute__((vector_size(16)));
__device__ __forceinline__ void async2(const __bf16* g, __bf16* l) {
  __builtin_amdgcn_global_load_async_to_lds_b128(
      (v4i_vs*)g, (__attribute__((address_space(3))) v4i_vs*)l, 0, 0);
  __builtin_amdgcn_global_load_async_to_lds_b128(
      (v4i_vs*)(g + 8), (__attribute__((address_space(3))) v4i_vs*)(l + 8), 0, 0);
}
#endif

// A-fragment for v_wmma_f32_16x16x32_bf16 from a row-major bf16 LDS row.
// lane element i holds K = 32*kstep + 16*half + i
__device__ __forceinline__ v16bf afrag(const __bf16* row, int hlf, int kstep) {
  const __bf16* p0 = row + kstep * 32 + hlf * 16;
  const __bf16* p1 = p0 + 8;
  v16bf a;
#pragma unroll
  for (int j = 0; j < 8; ++j) { a[j] = p0[j]; a[8 + j] = p1[j]; }
  return a;
}

// ---------------------------------------------------------------- utility kernels
__global__ void k_zero(float* p, int n) {
  int i = blockIdx.x * blockDim.x + threadIdx.x;
  if (i < n) p[i] = 0.f;
}
__global__ void k_count_edges(const int* __restrict__ dstp, float* ncnt) {
  int e = blockIdx.x * blockDim.x + threadIdx.x;
  if (e < EE) atomicAdd(&ncnt[dstp[e]], 1.f);
}
__global__ void k_count_nodes(const int* __restrict__ batch, float* gcnt) {
  int n = blockIdx.x * blockDim.x + threadIdx.x;
  if (n < NN) atomicAdd(&gcnt[batch[n]], 1.f);
}

// lin0: y = relu(x @ w + b), N x 64
__global__ __launch_bounds__(64) void k_lin0(const float* __restrict__ x,
                                             const float* __restrict__ w,
                                             const float* __restrict__ b,
                                             float* __restrict__ y) {
  __shared__ float row[64];
  int n = blockIdx.x, d = threadIdx.x;
  row[d] = x[(size_t)n * 64 + d];
  __syncthreads();
  float acc = b[d];
#pragma unroll 8
  for (int k = 0; k < 64; ++k) acc += row[k] * w[k * 64 + d];
  y[(size_t)n * 64 + d] = fmaxf(acc, 0.f);
}

// ---------------------------------------------------------------- weight repack (WMMA-B layout, bf16)
// B element at lane L, elem i: K = 32*kstep + 16*(L>>4) + i ; N(col) = 16*tile + (L&15)
__global__ void k_repack_w2(const float* __restrict__ w2, __bf16* __restrict__ dst) {
  int idx = blockIdx.x * blockDim.x + threadIdx.x;
  if (idx >= 3 * 256 * 2 * 32 * 16) return;
  int i = idx & 15, lane = (idx >> 4) & 31, s = (idx >> 9) & 1, t = (idx >> 10) & 255, l = idx >> 18;
  int K = s * 32 + (lane >> 4) * 16 + i;
  int c = t * 16 + (lane & 15);
  dst[idx] = (__bf16)w2[((size_t)l * 64 + K) * 4096 + c];
}
__global__ void k_repack_w1(const float* __restrict__ w1, __bf16* __restrict__ dst) {
  int idx = blockIdx.x * blockDim.x + threadIdx.x;
  if (idx >= 3 * 4 * 32 * 16) return;
  int i = idx & 15, lane = (idx >> 4) & 31, ot = (idx >> 9) & 3, l = idx >> 11;
  int K = (lane >> 4) * 16 + i, c = ot * 16 + (lane & 15);
  dst[idx] = (K < 16) ? (__bf16)w1[((size_t)l * 16 + K) * 64 + c] : (__bf16)0.f;
}
__global__ void k_repack_b2(const float* __restrict__ b2, __bf16* __restrict__ dst) {
  int idx = blockIdx.x * blockDim.x + threadIdx.x;
  if (idx >= 3 * 4 * 2 * 32 * 16) return;
  int i = idx & 15, lane = (idx >> 4) & 31, s = (idx >> 9) & 1, ot = (idx >> 10) & 3, l = idx >> 12;
  int K = s * 32 + (lane >> 4) * 16 + i, c = ot * 16 + (lane & 15);
  dst[idx] = (__bf16)b2[(size_t)l * 4096 + (size_t)K * 64 + c];
}

// ---------------------------------------------------------------- fused edge-message kernel (the hot one)
// One wave = 16 edges; 8 waves/block share B-fragments through a triple-buffered
// LDS stage (8 KB per dz-chunk) filled by the gfx1250 async global->LDS engine.
// h = relu(ea@w1+b1) via WMMA; msg = sum_d x_d * (h@w2)[:, d*64+o] + x@b2mat,
// fully fused (W[e] never materialized). Scatter-add msg into agg[dst].
__global__ __launch_bounds__(256) void k_edge(
    const float* __restrict__ x, const int* __restrict__ eidx,
    const float* __restrict__ eattr, const float* __restrict__ b1v,
    const __bf16* __restrict__ w2p, const __bf16* __restrict__ w1p,
    const __bf16* __restrict__ b2p, float* __restrict__ agg) {
  __shared__ __attribute__((aligned(16))) __bf16 s_x[8][16][64];
  __shared__ __attribute__((aligned(16))) __bf16 s_h[8][16][64];  // first 1 KB also stages ea
  __shared__ __attribute__((aligned(16))) __bf16 s_B[3][4096];    // triple-buffered w2 chunks

  const int tt = threadIdx.x;
  const int lane = tt & 31;
  const int wv = tt >> 5;
  const int tile = blockIdx.x * 8 + wv;
  const int eb = tile * 16;
  const int hlf = lane >> 4;
  const int col = lane & 15;
  const int* srcp = eidx;
  const int* dstp = eidx + EE;

#if USE_ASYNC
  // kick off chunk 0 on the async copy engine
  async2(w2p + (size_t)tt * 16, &s_B[0][tt * 16]);
#else
  // cooperative: w2 chunk 0 -> LDS buf0; prefetch chunk 1 into regs
  {
    v16bf c0 = *(const v16bf*)(w2p + (size_t)tt * 16);
    *(v16bf*)(&s_B[0][tt * 16]) = c0;
  }
  v16bf pf = *(const v16bf*)(w2p + 4096 + (size_t)tt * 16);
#endif

  __bf16* ea = &s_h[wv][0][0];  // aliased [16][32] staging area (consumed before h stores)
  {  // stage edge_attr (16x16), zero-pad K to 32
    int row = lane >> 1, c0 = (lane & 1) * 8;
    const float* p = eattr + (size_t)(eb + row) * 16 + c0;
#pragma unroll
    for (int j = 0; j < 8; ++j) {
      ea[row * 32 + c0 + j] = (__bf16)p[j];
      ea[row * 32 + 16 + c0 + j] = (__bf16)0.f;
    }
  }
  {  // stage gathered x[src] rows (16x64)
    int s = srcp[eb + col];
    const float* p = x + (size_t)s * 64 + hlf * 32;
#pragma unroll
    for (int j = 0; j < 32; ++j) s_x[wv][col][hlf * 32 + j] = (__bf16)p[j];
  }
  __syncthreads();

  v8f zz = {};
  {  // h = relu(ea @ w1 + b1) -> LDS (bf16); ea frag is in regs before h overwrites LDS
    v16bf eaA = afrag(ea + col * 32, hlf, 0);
#pragma unroll
    for (int ot = 0; ot < 4; ++ot) {
      v16bf bw = *(const v16bf*)(w1p + ((size_t)(ot * 32 + lane) << 4));
      v8f hc = __builtin_amdgcn_wmma_f32_16x16x32_bf16(false, eaA, false, bw,
                                                       (short)0, zz, false, false);
      float bias = b1v[ot * 16 + col];
#pragma unroll
      for (int r = 0; r < 8; ++r) {
        float v = hc[r] + bias;
        s_h[wv][r + 8 * hlf][ot * 16 + col] = (__bf16)(v > 0.f ? v : 0.f);
      }
    }
  }
  __syncthreads();

  v16bf hA0 = afrag(&s_h[wv][col][0], hlf, 0);
  v16bf hA1 = afrag(&s_h[wv][col][0], hlf, 1);
  v16bf xA0 = afrag(&s_x[wv][col][0], hlf, 0);
  v16bf xA1 = afrag(&s_x[wv][col][0], hlf, 1);

  // msg accumulators (16 edges x 64 outs as 4 C-tiles), seeded with x @ b2mat
  v8f msg[4];
#pragma unroll
  for (int ot = 0; ot < 4; ++ot) {
    v16bf B0 = *(const v16bf*)(b2p + ((size_t)((ot * 2 + 0) * 32 + lane) << 4));
    v16bf B1 = *(const v16bf*)(b2p + ((size_t)((ot * 2 + 1) * 32 + lane) << 4));
    v8f c = __builtin_amdgcn_wmma_f32_16x16x32_bf16(false, xA0, false, B0,
                                                    (short)0, zz, false, false);
    msg[ot] = __builtin_amdgcn_wmma_f32_16x16x32_bf16(false, xA1, false, B1,
                                                      (short)0, c, false, false);
  }

  // main contraction: Z = H @ w2 streamed in 16-col tiles through LDS; fold Z*x_d into msg
  int bcur = 0, bnxt = 1;
  for (int dz = 0; dz < 64; ++dz) {
#if USE_ASYNC
    // distance-1 prefetch of chunk dz+1 (tail re-issues chunk 63 so the wait
    // immediate stays uniform: own newest 2 ops are chunk dz+1 => chunk dz done)
    int ch = dz + 1;
    if (ch > 63) ch = 63;
    async2(w2p + (size_t)ch * 4096 + (size_t)tt * 16, &s_B[bnxt][tt * 16]);
    __builtin_amdgcn_s_wait_asynccnt(2);
#else
    if (dz < 63) *(v16bf*)(&s_B[bnxt][tt * 16]) = pf;                       // chunk dz+1
    if (dz < 62) pf = *(const v16bf*)(w2p + (size_t)(dz + 2) * 4096 + (size_t)tt * 16);
#endif
    __syncthreads();
    float xs[8];
#pragma unroll
    for (int r = 0; r < 8; ++r) xs[r] = (float)s_x[wv][r + 8 * hlf][dz];
#pragma unroll
    for (int ot = 0; ot < 4; ++ot) {
      const __bf16* pb = &s_B[bcur][((ot * 2) * 32 + lane) * 16];
      v16bf B0 = *(const v16bf*)pb;
      v16bf B1 = *(const v16bf*)(pb + 512);
      v8f z = __builtin_amdgcn_wmma_f32_16x16x32_bf16(false, hA0, false, B0,
                                                      (short)0, zz, false, false);
      z = __builtin_amdgcn_wmma_f32_16x16x32_bf16(false, hA1, false, B1,
                                                  (short)0, z, false, false);
#pragma unroll
      for (int r = 0; r < 8; ++r) msg[ot][r] += z[r] * xs[r];
    }
    bcur = bnxt;
    bnxt = (bnxt == 2) ? 0 : bnxt + 1;
  }

  // scatter-add to agg[dst]
#pragma unroll
  for (int r = 0; r < 8; ++r) {
    int dn = dstp[eb + r + 8 * hlf];
#pragma unroll
    for (int ot = 0; ot < 4; ++ot)
      atomicAdd(&agg[(size_t)dn * 64 + ot * 16 + col], msg[ot][r]);
  }
}

// ---------------------------------------------------------------- per-layer glue
__global__ __launch_bounds__(64) void k_combine(
    const float* __restrict__ x, const float* __restrict__ agg,
    const float* __restrict__ ncnt, const float* __restrict__ rootw,
    const float* __restrict__ cb, const int* __restrict__ batch,
    float* __restrict__ xc, float* __restrict__ gsum) {
  __shared__ float row[64];
  int n = blockIdx.x, d = threadIdx.x;
  row[d] = x[(size_t)n * 64 + d];
  __syncthreads();
  float acc = agg[(size_t)n * 64 + d] / fmaxf(ncnt[n], 1.f) + cb[d];
#pragma unroll 8
  for (int k = 0; k < 64; ++k) acc += row[k] * rootw[k * 64 + d];
  acc = fmaxf(acc, 0.f);
  xc[(size_t)n * 64 + d] = acc;
  atomicAdd(&gsum[(size_t)batch[n] * 64 + d], acc);
}
__global__ void k_segdiv(float* out, const float* sum, const float* cnt, int n) {
  int i = blockIdx.x * blockDim.x + threadIdx.x;
  if (i < n) out[i] = sum[i] / fmaxf(cnt[i >> 6], 1.f);
}
__global__ void k_sub(float* xc, const float* mean, const float* sc,
                      const int* batch, float* gvar) {
  int i = blockIdx.x * blockDim.x + threadIdx.x;
  if (i >= NN * 64) return;
  int n = i >> 6, d = i & 63, g = batch[n];
  float s = xc[i] - sc[d] * mean[g * 64 + d];
  xc[i] = s;
  atomicAdd(&gvar[g * 64 + d], s * s);
}
__global__ void k_norm(float* xo, const float* sub, const float* var,
                       const float* w, const float* b, const float* res,
                       const int* batch) {
  int i = blockIdx.x * blockDim.x + threadIdx.x;
  if (i >= NN * 64) return;
  int n = i >> 6, d = i & 63, g = batch[n];
  xo[i] = w[d] * sub[i] * rsqrtf(var[g * 64 + d] + EPSV) + b[d] + res[i];
}

// ---------------------------------------------------------------- Set2Set + head
__global__ __launch_bounds__(64) void k_lstm(const float* __restrict__ qstar,
                                             float* hh, float* cc,
                                             const float* __restrict__ wih,
                                             const float* __restrict__ whh,
                                             const float* __restrict__ bih,
                                             const float* __restrict__ bhh) {
  __shared__ float qs[128];
  __shared__ float hv[64];
  int g = blockIdx.x, d = threadIdx.x;
  qs[d] = qstar[g * 128 + d];
  qs[64 + d] = qstar[g * 128 + 64 + d];
  hv[d] = hh[g * 64 + d];
  __syncthreads();
  float gate[4];
#pragma unroll
  for (int gi = 0; gi < 4; ++gi) {
    int row = gi * 64 + d;
    float acc = bih[row] + bhh[row];
    for (int k = 0; k < 128; ++k) acc += qs[k] * wih[row * 128 + k];
    for (int k = 0; k < 64; ++k) acc += hv[k] * whh[row * 64 + k];
    gate[gi] = acc;
  }
  float ig = 1.f / (1.f + expf(-gate[0]));
  float fg = 1.f / (1.f + expf(-gate[1]));
  float gg = tanhf(gate[2]);
  float og = 1.f / (1.f + expf(-gate[3]));
  float c = fg * cc[g * 64 + d] + ig * gg;
  cc[g * 64 + d] = c;
  hh[g * 64 + d] = og * tanhf(c);
}
__global__ void k_att_e(const float* __restrict__ x, const float* __restrict__ hh,
                        const int* __restrict__ batch, float* ev, unsigned* emax) {
  int n = blockIdx.x * blockDim.x + threadIdx.x;
  if (n >= NN) return;
  int g = batch[n];
  float acc = 0.f;
#pragma unroll 8
  for (int d = 0; d < 64; ++d) acc += x[(size_t)n * 64 + d] * hh[g * 64 + d];
  ev[n] = acc;
  atomicMax(&emax[g], f32_ord(acc));
}
__global__ void k_att_exp(const float* ev, const unsigned* emax,
                          const int* batch, float* ex, float* den) {
  int n = blockIdx.x * blockDim.x + threadIdx.x;
  if (n >= NN) return;
  int g = batch[n];
  float e = expf(ev[n] - ord_f32(emax[g]));
  ex[n] = e;
  atomicAdd(&den[g], e);
}
__global__ void k_att_r(const float* ex, const float* den, const float* x,
                        const int* batch, float* rv) {
  int i = blockIdx.x * blockDim.x + threadIdx.x;
  if (i >= NN * 64) return;
  int n = i >> 6, d = i & 63, g = batch[n];
  float a = ex[n] / den[g];
  atomicAdd(&rv[g * 64 + d], a * x[i]);
}
__global__ void k_qstar(const float* hh, const float* rv, float* qs) {
  int i = blockIdx.x * blockDim.x + threadIdx.x;
  if (i >= GG * 128) return;
  int g = i >> 7, j = i & 127;
  qs[i] = (j < 64) ? hh[g * 64 + j] : rv[g * 64 + j - 64];
}
__global__ __launch_bounds__(64) void k_head(const float* __restrict__ qstar,
                                             const float* __restrict__ w1,
                                             const float* __restrict__ b1,
                                             const float* __restrict__ w2,
                                             const float* __restrict__ b2,
                                             float* out) {
  __shared__ float qs[128];
  __shared__ float t1[64];
  int g = blockIdx.x, d = threadIdx.x;
  qs[d] = qstar[g * 128 + d];
  qs[64 + d] = qstar[g * 128 + 64 + d];
  __syncthreads();
  float acc = b1[d];
  for (int k = 0; k < 128; ++k) acc += qs[k] * w1[k * 64 + d];
  t1[d] = fmaxf(acc, 0.f);
  __syncthreads();
  if (d == 0) {
    float o = b2[0];
    for (int k = 0; k < 64; ++k) o += t1[k] * w2[k];
    out[g] = o;
  }
}

// ---------------------------------------------------------------- host launcher
extern "C" void kernel_launch(void* const* d_in, const int* in_sizes, int n_in,
                              void* d_out, int out_size, void* d_ws, size_t ws_size,
                              hipStream_t stream) {
  (void)in_sizes; (void)n_in; (void)out_size; (void)ws_size;
  const float* x_in = (const float*)d_in[0];
  const int* eidx = (const int*)d_in[1];
  const float* eattr = (const float*)d_in[2];
  const int* batch = (const int*)d_in[3];
  const float* lin0_w = (const float*)d_in[4];
  const float* lin0_b = (const float*)d_in[5];
  const float* mlp_w1 = (const float*)d_in[6];
  const float* mlp_b1 = (const float*)d_in[7];
  const float* mlp_w2 = (const float*)d_in[8];
  const float* mlp_b2 = (const float*)d_in[9];
  const float* root_w = (const float*)d_in[10];
  const float* conv_b = (const float*)d_in[11];
  const float* gn_w = (const float*)d_in[12];
  const float* gn_b = (const float*)d_in[13];
  const float* gn_sc = (const float*)d_in[14];
  const float* wih = (const float*)d_in[15];
  const float* whh = (const float*)d_in[16];
  const float* bih = (const float*)d_in[17];
  const float* bhh = (const float*)d_in[18];
  const float* hw1 = (const float*)d_in[19];
  const float* hb1 = (const float*)d_in[20];
  const float* hw2 = (const float*)d_in[21];
  const float* hb2 = (const float*)d_in[22];
  float* out = (float*)d_out;
  char* ws = (char*)d_ws;

  constexpr size_t OFF_XA = 0;
  constexpr size_t OFF_XB = OFF_XA + (size_t)NN * 64 * 4;
  constexpr size_t OFF_AGG = OFF_XB + (size_t)NN * 64 * 4;
  constexpr size_t OFF_XC = OFF_AGG + (size_t)NN * 64 * 4;
  constexpr size_t OFF_NCNT = OFF_XC + (size_t)NN * 64 * 4;
  constexpr size_t OFF_GCNT = OFF_NCNT + (size_t)NN * 4;
  constexpr size_t OFF_GSUM = OFF_GCNT + 4096;
  constexpr size_t OFF_GVAR = OFF_GSUM + (size_t)GG * 64 * 4;
  constexpr size_t OFF_GMEAN = OFF_GVAR + (size_t)GG * 64 * 4;
  constexpr size_t OFF_GVARM = OFF_GMEAN + (size_t)GG * 64 * 4;
  constexpr size_t OFF_W2P = OFF_GVARM + (size_t)GG * 64 * 4;
  constexpr size_t OFF_W1P = OFF_W2P + (size_t)3 * 256 * 2 * 32 * 16 * 2;
  constexpr size_t OFF_B2P = OFF_W1P + (size_t)3 * 4 * 32 * 16 * 2;
  constexpr size_t OFF_HH = OFF_B2P + (size_t)3 * 4 * 2 * 32 * 16 * 2;
  constexpr size_t OFF_CC = OFF_HH + (size_t)GG * 64 * 4;
  constexpr size_t OFF_QS = OFF_CC + (size_t)GG * 64 * 4;
  constexpr size_t OFF_RV = OFF_QS + (size_t)GG * 128 * 4;
  constexpr size_t OFF_EV = OFF_RV + (size_t)GG * 64 * 4;
  constexpr size_t OFF_EX = OFF_EV + (size_t)NN * 4;
  constexpr size_t OFF_EM = OFF_EX + (size_t)NN * 4;
  constexpr size_t OFF_DEN = OFF_EM + 4096;

  float* xa = (float*)(ws + OFF_XA);
  float* xb = (float*)(ws + OFF_XB);
  float* aggp = (float*)(ws + OFF_AGG);
  float* xcp = (float*)(ws + OFF_XC);
  float* ncntp = (float*)(ws + OFF_NCNT);
  float* gcntp = (float*)(ws + OFF_GCNT);
  float* gsump = (float*)(ws + OFF_GSUM);
  float* gvarp = (float*)(ws + OFF_GVAR);
  float* gmeanp = (float*)(ws + OFF_GMEAN);
  float* gvarmp = (float*)(ws + OFF_GVARM);
  __bf16* w2p = (__bf16*)(ws + OFF_W2P);
  __bf16* w1p = (__bf16*)(ws + OFF_W1P);
  __bf16* b2p = (__bf16*)(ws + OFF_B2P);
  float* hhp = (float*)(ws + OFF_HH);
  float* ccp = (float*)(ws + OFF_CC);
  float* qsp = (float*)(ws + OFF_QS);
  float* rvp = (float*)(ws + OFF_RV);
  float* evp = (float*)(ws + OFF_EV);
  float* exvp = (float*)(ws + OFF_EX);
  unsigned* emaxp = (unsigned*)(ws + OFF_EM);
  float* denp = (float*)(ws + OFF_DEN);

  // ---- one-time (per call) setup
  k_zero<<<68, 256, 0, stream>>>(ncntp, NN + 1024);  // ncnt + gcnt region
  k_count_edges<<<EE / 256, 256, 0, stream>>>(eidx + EE, ncntp);
  k_count_nodes<<<NN / 256, 256, 0, stream>>>(batch, gcntp);
  k_lin0<<<NN, 64, 0, stream>>>(x_in, lin0_w, lin0_b, xa);
  k_repack_w2<<<3072, 256, 0, stream>>>(mlp_w2, w2p);
  k_repack_w1<<<24, 256, 0, stream>>>(mlp_w1, w1p);
  k_repack_b2<<<48, 256, 0, stream>>>(mlp_b2, b2p);

  // ---- layers
  float* xcur = xa;
  float* xnxt = xb;
  for (int l = 0; l < LL; ++l) {
    k_zero<<<(NN * 64) / 256, 256, 0, stream>>>(aggp, NN * 64);
    k_zero<<<(GG * 64 * 2) / 256, 256, 0, stream>>>(gsump, GG * 64 * 2);  // gsum+gvar
    k_edge<<<512, 256, 0, stream>>>(xcur, eidx, eattr, mlp_b1 + l * 64,
                                    w2p + (size_t)l * 262144,
                                    w1p + (size_t)l * 2048,
                                    b2p + (size_t)l * 4096, aggp);
    k_combine<<<NN, 64, 0, stream>>>(xcur, aggp, ncntp, root_w + l * 4096,
                                     conv_b + l * 64, batch, xcp, gsump);
    k_segdiv<<<32, 256, 0, stream>>>(gmeanp, gsump, gcntp, GG * 64);
    k_sub<<<(NN * 64) / 256, 256, 0, stream>>>(xcp, gmeanp, gn_sc + l * 64, batch, gvarp);
    k_segdiv<<<32, 256, 0, stream>>>(gvarmp, gvarp, gcntp, GG * 64);
    k_norm<<<(NN * 64) / 256, 256, 0, stream>>>(xnxt, xcp, gvarmp, gn_w + l * 64,
                                                gn_b + l * 64, xcur, batch);
    float* t = xcur; xcur = xnxt; xnxt = t;
  }

  // ---- Set2Set
  k_zero<<<160, 256, 0, stream>>>(hhp, 40960);  // hh, cc, qstar, r
  for (int s = 0; s < 3; ++s) {
    k_lstm<<<GG, 64, 0, stream>>>(qsp, hhp, ccp, wih, whh, bih, bhh);
    k_zero<<<32, 256, 0, stream>>>(rvp, GG * 64);
    k_zero<<<5, 256, 0, stream>>>((float*)(ws + OFF_EM), 1152);  // emax + denom
    k_att_e<<<NN / 256, 256, 0, stream>>>(xcur, hhp, batch, evp, emaxp);
    k_att_exp<<<NN / 256, 256, 0, stream>>>(evp, emaxp, batch, exvp, denp);
    k_att_r<<<(NN * 64) / 256, 256, 0, stream>>>(exvp, denp, xcur, batch, rvp);
    k_qstar<<<(GG * 128) / 256, 256, 0, stream>>>(hhp, rvp, qsp);
  }
  k_head<<<GG, 64, 0, stream>>>(qsp, hw1, hb1, hw2, hb2, out);
}
